// SurpriseBasedMemory_43276090474628
// MI455X (gfx1250) — matrix operand order repaired and checked by reading, not compiled, not run on hardware
//
#include <hip/hip_runtime.h>
#include <math.h>

typedef float v8f  __attribute__((ext_vector_type(8)));
typedef __bf16 v8bf  __attribute__((ext_vector_type(8)));
typedef __bf16 v16bf __attribute__((ext_vector_type(16)));

#define H_DIM 1024
#define M_ROWS 4096
#define N_TOK 8192
#define TOPK 64
#define THRESH 0.5f
#define LN_EPS 1e-12f

// ---------------------------------------------------------------------------
// bf16 split helpers (round-to-nearest-even)
// ---------------------------------------------------------------------------
__device__ __forceinline__ unsigned short f32_to_bf16(float f) {
    unsigned int u = __float_as_uint(f);
    u += 0x7FFFu + ((u >> 16) & 1u);
    return (unsigned short)(u >> 16);
}
__device__ __forceinline__ float bf16_to_f32(unsigned short h) {
    return __uint_as_float(((unsigned int)h) << 16);
}
__device__ __forceinline__ void split_bf16(float f, unsigned short& hi, unsigned short& lo) {
    hi = f32_to_bf16(f);
    lo = f32_to_bf16(f - bf16_to_f32(hi));
}

// Fragment loads. 16-bit A 16x32 layout: lanes 0-15 hold K = {0..7, 16..23},
// lanes 16-31 hold K = {8..15, 24..31}  -> two b128 loads at +0 / +16 halves.
// 16-bit B 32x16 layout: same as A: lanes 0-15 hold K = {0..7, 16..23},
// lanes 16-31 hold K = {8..15, 24..31} -> two b128 loads at +0 / +16 halves.
__device__ __forceinline__ v16bf load_fragA(const unsigned short* p) {
    union { v16bf v; v8bf h[2]; } u;
    u.h[0] = *(const v8bf*)(p);
    u.h[1] = *(const v8bf*)(p + 16);
    return u.v;
}
__device__ __forceinline__ v16bf load_fragB(const unsigned short* p) {
    union { v16bf v; v8bf h[2]; } u;
    u.h[0] = *(const v8bf*)(p);
    u.h[1] = *(const v8bf*)(p + 16);
    return u.v;
}

// ---------------------------------------------------------------------------
// Kernel 1: LayerNorm + surprise; emits bf16 hi/lo planes of `norm`
// ---------------------------------------------------------------------------
__global__ __launch_bounds__(256)
void k_ln_surprise(const float* __restrict__ x,
                   const float* __restrict__ gamma,
                   const float* __restrict__ beta,
                   unsigned short* __restrict__ nhi,
                   unsigned short* __restrict__ nlo,
                   float* __restrict__ surprise)
{
    const int row = blockIdx.x;
    const int t = threadIdx.x;
    const float* xr = x + (size_t)row * H_DIM;

    float v[4];
    float s = 0.f, s2 = 0.f;
    #pragma unroll
    for (int i = 0; i < 4; ++i) {
        float u = xr[t + 256 * i];
        v[i] = u; s += u; s2 += u * u;
    }
    __shared__ float r1[256];
    __shared__ float r2[256];
    r1[t] = s; r2[t] = s2;
    __syncthreads();
    for (int off = 128; off > 0; off >>= 1) {
        if (t < off) { r1[t] += r1[t + off]; r2[t] += r2[t + off]; }
        __syncthreads();
    }
    const float mu  = r1[0] * (1.0f / H_DIM);
    const float var = fmaxf(r2[0] * (1.0f / H_DIM) - mu * mu, 0.0f);
    const float inv = 1.0f / sqrtf(var + LN_EPS);
    __syncthreads();

    float asum = 0.f;
    #pragma unroll
    for (int i = 0; i < 4; ++i) {
        int c = t + 256 * i;
        float nv = (v[i] - mu) * inv * gamma[c] + beta[c];
        unsigned short hi, lo;
        split_bf16(nv, hi, lo);
        nhi[(size_t)row * H_DIM + c] = hi;
        nlo[(size_t)row * H_DIM + c] = lo;
        asum += fabsf(nv);
    }
    r1[t] = asum;
    __syncthreads();
    for (int off = 128; off > 0; off >>= 1) {
        if (t < off) r1[t] += r1[t + off];
        __syncthreads();
    }
    if (t == 0) surprise[row] = r1[0] * (1.0f / H_DIM);
}

// ---------------------------------------------------------------------------
// Kernel 2: top-64 surprises (desc, ties -> lower index, jax top_k semantics)
// ---------------------------------------------------------------------------
__global__ __launch_bounds__(256)
void k_top_surprise(const float* __restrict__ s,
                    int* __restrict__ top_idx,
                    int* __restrict__ top_valid)
{
    __shared__ float vals[N_TOK];
    __shared__ float bv[256];
    __shared__ int   bi[256];
    const int t = threadIdx.x;
    for (int i = t; i < N_TOK; i += 256) vals[i] = s[i];
    __syncthreads();

    for (int k = 0; k < TOPK; ++k) {
        float mv = -INFINITY; int mi = -1;
        for (int i = t; i < N_TOK; i += 256) {
            float v = vals[i];
            if (v > mv) { mv = v; mi = i; }
        }
        bv[t] = mv; bi[t] = mi;
        __syncthreads();
        for (int off = 128; off > 0; off >>= 1) {
            if (t < off) {
                float ov = bv[t + off]; int oi = bi[t + off];
                if (ov > bv[t] || (ov == bv[t] && oi >= 0 && (bi[t] < 0 || oi < bi[t]))) {
                    bv[t] = ov; bi[t] = oi;
                }
            }
            __syncthreads();
        }
        if (t == 0) {
            top_idx[k]   = bi[0];
            top_valid[k] = (bv[0] > THRESH) ? 1 : 0;
            vals[bi[0]] = -INFINITY;
        }
        __syncthreads();
    }
}

// ---------------------------------------------------------------------------
// Kernel 3: 64 slots with smallest importance (asc, ties -> lower index)
// ---------------------------------------------------------------------------
__global__ __launch_bounds__(256)
void k_top_importance(const float* __restrict__ imp, int* __restrict__ slots)
{
    __shared__ float vals[M_ROWS];
    __shared__ float bv[256];
    __shared__ int   bi[256];
    const int t = threadIdx.x;
    for (int i = t; i < M_ROWS; i += 256) vals[i] = imp[i];
    __syncthreads();

    for (int k = 0; k < TOPK; ++k) {
        float mv = INFINITY; int mi = -1;
        for (int i = t; i < M_ROWS; i += 256) {
            float v = vals[i];
            if (v < mv) { mv = v; mi = i; }
        }
        bv[t] = mv; bi[t] = mi;
        __syncthreads();
        for (int off = 128; off > 0; off >>= 1) {
            if (t < off) {
                float ov = bv[t + off]; int oi = bi[t + off];
                if (ov < bv[t] || (ov == bv[t] && oi >= 0 && (bi[t] < 0 || oi < bi[t]))) {
                    bv[t] = ov; bi[t] = oi;
                }
            }
            __syncthreads();
        }
        if (t == 0) {
            slots[k] = bi[0];
            vals[bi[0]] = INFINITY;
        }
        __syncthreads();
    }
}

// ---------------------------------------------------------------------------
// Kernel 4: split a f32 matrix into bf16 hi/lo planes (for Wq, Wo)
// ---------------------------------------------------------------------------
__global__ __launch_bounds__(256)
void k_split_plane(const float* __restrict__ src,
                   unsigned short* __restrict__ hi,
                   unsigned short* __restrict__ lo, int n)
{
    int i = blockIdx.x * 256 + threadIdx.x;
    if (i < n) {
        unsigned short h, l;
        split_bf16(src[i], h, l);
        hi[i] = h; lo[i] = l;
    }
}

// ---------------------------------------------------------------------------
// Kernel 5: build updated memory as bf16 hi/lo planes
// ---------------------------------------------------------------------------
__global__ __launch_bounds__(256)
void k_build_mem(const float* __restrict__ memory,
                 const unsigned short* __restrict__ nhi,
                 const unsigned short* __restrict__ nlo,
                 const int* __restrict__ slots,
                 const int* __restrict__ top_idx,
                 const int* __restrict__ top_valid,
                 unsigned short* __restrict__ mhi,
                 unsigned short* __restrict__ mlo)
{
    const int r = blockIdx.x;
    __shared__ int src;
    if (threadIdx.x == 0) {
        int sidx = -1;
        for (int k = 0; k < TOPK; ++k)
            if (slots[k] == r) { if (top_valid[k]) sidx = top_idx[k]; break; }
        src = sidx;
    }
    __syncthreads();
    unsigned short* th = mhi + (size_t)r * H_DIM;
    unsigned short* tl = mlo + (size_t)r * H_DIM;
    if (src >= 0) {
        const unsigned short* fh = nhi + (size_t)src * H_DIM;
        const unsigned short* fl = nlo + (size_t)src * H_DIM;
        for (int i = threadIdx.x; i < H_DIM; i += 256) { th[i] = fh[i]; tl[i] = fl[i]; }
    } else {
        const float* fm = memory + (size_t)r * H_DIM;
        for (int i = threadIdx.x; i < H_DIM; i += 256) {
            unsigned short h, l;
            split_bf16(fm[i], h, l);
            th[i] = h; tl[i] = l;
        }
    }
}

// ---------------------------------------------------------------------------
// Kernel 6: LDS-tiled transpose of bf16 planes ([rows,cols] -> [cols,rows])
// ---------------------------------------------------------------------------
__global__ __launch_bounds__(1024)
void k_transpose_planes(const unsigned short* __restrict__ shi,
                        const unsigned short* __restrict__ slo,
                        unsigned short* __restrict__ dhi,
                        unsigned short* __restrict__ dlo,
                        int rows, int cols)
{
    __shared__ unsigned short th[32][33];
    __shared__ unsigned short tl[32][33];
    const int c0 = blockIdx.x * 32, r0 = blockIdx.y * 32;
    const int tx = threadIdx.x, ty = threadIdx.y;
    th[ty][tx] = shi[(size_t)(r0 + ty) * cols + c0 + tx];
    tl[ty][tx] = slo[(size_t)(r0 + ty) * cols + c0 + tx];
    __syncthreads();
    dhi[(size_t)(c0 + ty) * rows + r0 + tx] = th[tx][ty];
    dlo[(size_t)(c0 + ty) * rows + r0 + tx] = tl[tx][ty];
}

// ---------------------------------------------------------------------------
// 3xBF16 split-GEMM: C[M,N] = A[M,K] * B[N,K]^T (+bias) (+residual)
//   A*B ~= Ahi*Bhi + Alo*Bhi + Ahi*Blo  (f32 accumulation; ~2^-16 rel. error)
// One wave per block, 64x64 tile, 16 accumulators, 48 WMMA per K-step of 32.
// ---------------------------------------------------------------------------
template <bool HAS_BIAS, bool HAS_RES, bool SPLIT_OUT>
__global__ __launch_bounds__(32)
void wmma_gemm_bf16x3(const unsigned short* __restrict__ Ahi, const unsigned short* __restrict__ Alo,
                      const unsigned short* __restrict__ Bhi, const unsigned short* __restrict__ Blo,
                      const float* __restrict__ bias, const float* __restrict__ res,
                      float* __restrict__ Cf,
                      unsigned short* __restrict__ Chi, unsigned short* __restrict__ Clo,
                      int M, int N, int K)
{
    const int n0 = blockIdx.x * 64;
    const int m0 = blockIdx.y * 64;
    const int lane = threadIdx.x & 31;
    const int lm = lane & 15;
    const int lh = lane >> 4;

    v8f acc[4][4] = {};
    const unsigned short *Ah[4], *Al[4], *Bh[4], *Bl[4];
    #pragma unroll
    for (int i = 0; i < 4; ++i) {
        const size_t ra = (size_t)(m0 + 16 * i + lm) * K + 8 * lh;
        Ah[i] = Ahi + ra; Al[i] = Alo + ra;
    }
    #pragma unroll
    for (int j = 0; j < 4; ++j) {
        const size_t rb = (size_t)(n0 + 16 * j + lm) * K + 8 * lh;
        Bh[j] = Bhi + rb; Bl[j] = Blo + rb;
    }

    for (int k = 0; k < K; k += 32) {
        v16bf ah[4], al[4], bh[4], bl[4];
        #pragma unroll
        for (int i = 0; i < 4; ++i) { ah[i] = load_fragA(Ah[i] + k); al[i] = load_fragA(Al[i] + k); }
        #pragma unroll
        for (int j = 0; j < 4; ++j) { bh[j] = load_fragB(Bh[j] + k); bl[j] = load_fragB(Bl[j] + k); }
        #pragma unroll
        for (int i = 0; i < 4; ++i) {
            #pragma unroll
            for (int j = 0; j < 4; ++j) {
                acc[i][j] = __builtin_amdgcn_wmma_f32_16x16x32_bf16(
                    false, ah[i], false, bh[j], (short)0, acc[i][j], false, false);
                acc[i][j] = __builtin_amdgcn_wmma_f32_16x16x32_bf16(
                    false, al[i], false, bh[j], (short)0, acc[i][j], false, false);
                acc[i][j] = __builtin_amdgcn_wmma_f32_16x16x32_bf16(
                    false, ah[i], false, bl[j], (short)0, acc[i][j], false, false);
            }
        }
    }

    #pragma unroll
    for (int i = 0; i < 4; ++i) {
        #pragma unroll
        for (int j = 0; j < 4; ++j) {
            const int col = n0 + 16 * j + lm;
            const float bvv = HAS_BIAS ? bias[col] : 0.0f;
            #pragma unroll
            for (int r = 0; r < 8; ++r) {
                const int row = m0 + 16 * i + r + 8 * lh;
                float v = acc[i][j][r] + bvv;
                if (HAS_RES) v += res[(size_t)row * N + col];
                if (SPLIT_OUT) {
                    unsigned short h, l;
                    split_bf16(v, h, l);
                    Chi[(size_t)row * N + col] = h;
                    Clo[(size_t)row * N + col] = l;
                } else {
                    Cf[(size_t)row * N + col] = v;
                }
            }
        }
    }
}

// ---------------------------------------------------------------------------
// Kernel: row softmax of f32 sim -> bf16 hi/lo attention planes
// ---------------------------------------------------------------------------
__global__ __launch_bounds__(256)
void k_softmax(const float* __restrict__ sim,
               unsigned short* __restrict__ ahi,
               unsigned short* __restrict__ alo)
{
    const int row = blockIdx.x;
    const int t = threadIdx.x;
    const float* p = sim + (size_t)row * M_ROWS;

    __shared__ float red[256];
    float mx = -INFINITY;
    for (int i = t; i < M_ROWS; i += 256) mx = fmaxf(mx, p[i]);
    red[t] = mx;
    __syncthreads();
    for (int off = 128; off > 0; off >>= 1) {
        if (t < off) red[t] = fmaxf(red[t], red[t + off]);
        __syncthreads();
    }
    mx = red[0];
    __syncthreads();

    float s = 0.f;
    for (int i = t; i < M_ROWS; i += 256) s += expf(p[i] - mx);
    red[t] = s;
    __syncthreads();
    for (int off = 128; off > 0; off >>= 1) {
        if (t < off) red[t] += red[t + off];
        __syncthreads();
    }
    const float inv = 1.0f / red[0];

    for (int i = t; i < M_ROWS; i += 256) {
        float e = expf(p[i] - mx) * inv;
        unsigned short h, l;
        split_bf16(e, h, l);
        ahi[(size_t)row * M_ROWS + i] = h;
        alo[(size_t)row * M_ROWS + i] = l;
    }
}

// ---------------------------------------------------------------------------
// launch
// ---------------------------------------------------------------------------
extern "C" void kernel_launch(void* const* d_in, const int* in_sizes, int n_in,
                              void* d_out, int out_size, void* d_ws, size_t ws_size,
                              hipStream_t stream)
{
    const float* hidden = (const float*)d_in[0];
    const float* gamma  = (const float*)d_in[1];
    const float* beta   = (const float*)d_in[2];
    const float* Wq     = (const float*)d_in[3];
    const float* bq     = (const float*)d_in[4];
    const float* Wo     = (const float*)d_in[5];
    const float* bo     = (const float*)d_in[6];
    const float* memory = (const float*)d_in[7];
    const float* imp    = (const float*)d_in[8];
    float* out = (float*)d_out;

    // workspace bump allocator (256B aligned)
    char* base = (char*)d_ws;
    size_t off = 0;
    auto alloc = [&](size_t bytes) -> char* {
        char* p = base + off;
        off += (bytes + 255) & ~(size_t)255;
        return p;
    };
    const size_t NH = (size_t)N_TOK * H_DIM;   // 8.39M
    const size_t MH = (size_t)M_ROWS * H_DIM;  // 4.19M
    const size_t NM = (size_t)N_TOK * M_ROWS;  // 33.55M

    unsigned short* norm_hi  = (unsigned short*)alloc(NH * 2);
    unsigned short* norm_lo  = (unsigned short*)alloc(NH * 2);
    unsigned short* q_hi     = (unsigned short*)alloc(NH * 2);
    unsigned short* q_lo     = (unsigned short*)alloc(NH * 2);
    unsigned short* Wq_hi    = (unsigned short*)alloc((size_t)H_DIM * H_DIM * 2);
    unsigned short* Wq_lo    = (unsigned short*)alloc((size_t)H_DIM * H_DIM * 2);
    unsigned short* Wo_hi    = (unsigned short*)alloc((size_t)H_DIM * H_DIM * 2);
    unsigned short* Wo_lo    = (unsigned short*)alloc((size_t)H_DIM * H_DIM * 2);
    unsigned short* mem2_hi  = (unsigned short*)alloc(MH * 2);
    unsigned short* mem2_lo  = (unsigned short*)alloc(MH * 2);
    unsigned short* mem2T_hi = (unsigned short*)alloc(MH * 2);
    unsigned short* mem2T_lo = (unsigned short*)alloc(MH * 2);
    float*          sim      = (float*)alloc(NM * 4);
    unsigned short* attn_hi  = (unsigned short*)alloc(NM * 2);
    unsigned short* attn_lo  = (unsigned short*)alloc(NM * 2);
    float*          surprise = (float*)alloc(N_TOK * 4);
    int*            top_idx  = (int*)alloc(TOPK * 4);
    int*            top_vld  = (int*)alloc(TOPK * 4);
    int*            slots    = (int*)alloc(TOPK * 4);
    // retrieved planes reuse the (dead) norm planes
    unsigned short* retr_hi = norm_hi;
    unsigned short* retr_lo = norm_lo;

    // 1) LayerNorm -> norm planes + surprise
    k_ln_surprise<<<N_TOK, 256, 0, stream>>>(hidden, gamma, beta, norm_hi, norm_lo, surprise);

    // 2) top-K selections
    k_top_surprise<<<1, 256, 0, stream>>>(surprise, top_idx, top_vld);
    k_top_importance<<<1, 256, 0, stream>>>(imp, slots);

    // 3) weight planes
    k_split_plane<<<(H_DIM * H_DIM) / 256, 256, 0, stream>>>(Wq, Wq_hi, Wq_lo, H_DIM * H_DIM);
    k_split_plane<<<(H_DIM * H_DIM) / 256, 256, 0, stream>>>(Wo, Wo_hi, Wo_lo, H_DIM * H_DIM);

    // 4) updated memory planes + K-major transpose for the PV GEMM
    k_build_mem<<<M_ROWS, 256, 0, stream>>>(memory, norm_hi, norm_lo, slots, top_idx, top_vld,
                                            mem2_hi, mem2_lo);
    k_transpose_planes<<<dim3(H_DIM / 32, M_ROWS / 32), dim3(32, 32), 0, stream>>>(
        mem2_hi, mem2_lo, mem2T_hi, mem2T_lo, M_ROWS, H_DIM);

    // 5) q = norm @ Wq^T + bq  -> q planes
    wmma_gemm_bf16x3<true, false, true><<<dim3(H_DIM / 64, N_TOK / 64), 32, 0, stream>>>(
        norm_hi, norm_lo, Wq_hi, Wq_lo, bq, nullptr, nullptr, q_hi, q_lo,
        N_TOK, H_DIM, H_DIM);

    // 6) sim = q @ mem2^T (f32, L2-resident 134MB)
    wmma_gemm_bf16x3<false, false, false><<<dim3(M_ROWS / 64, N_TOK / 64), 32, 0, stream>>>(
        q_hi, q_lo, mem2_hi, mem2_lo, nullptr, nullptr, sim, nullptr, nullptr,
        N_TOK, M_ROWS, H_DIM);

    // 7) softmax rows -> attention planes
    k_softmax<<<N_TOK, 256, 0, stream>>>(sim, attn_hi, attn_lo);

    // 8) retrieved = attn @ mem2  (Bt form via mem2T) -> retr planes
    wmma_gemm_bf16x3<false, false, true><<<dim3(H_DIM / 64, N_TOK / 64), 32, 0, stream>>>(
        attn_hi, attn_lo, mem2T_hi, mem2T_lo, nullptr, nullptr, nullptr, retr_hi, retr_lo,
        N_TOK, H_DIM, M_ROWS);

    // 9) out = hidden + retrieved @ Wo^T + bo
    wmma_gemm_bf16x3<true, true, false><<<dim3(H_DIM / 64, N_TOK / 64), 32, 0, stream>>>(
        retr_hi, retr_lo, Wo_hi, Wo_lo, bo, hidden, out, nullptr, nullptr,
        N_TOK, H_DIM, H_DIM);
}
